// KhmerRNN_52037823758736
// MI455X (gfx1250) — matrix operand order repaired and checked by reading, not compiled, not run on hardware
//
#include <hip/hip_runtime.h>
#include <hip/hip_bf16.h>
#include <math.h>

#define Vv 30000
#define Ee 128
#define Hh 256
#define NTAG 2
#define Bb 32
#define Tt 512

typedef __attribute__((ext_vector_type(16))) __bf16       v16bf;
typedef __attribute__((ext_vector_type(8)))  float        v8f;
typedef __attribute__((ext_vector_type(4)))  unsigned int u32x4;

union Frag16 { v16bf v; u32x4 q[2]; };

__device__ __forceinline__ unsigned short f32_to_bf16(float f) {
  union { float f; unsigned int u; } c; c.f = f;
  unsigned int u = c.u;
  u += 0x7FFFu + ((u >> 16) & 1u);          // round-to-nearest-even
  return (unsigned short)(u >> 16);
}

// ---------------------------------------------------------------------------
// Embedding gather -> bf16 activations [B, T, E]
// ---------------------------------------------------------------------------
__global__ void embed_kernel(const int* __restrict__ x, const float* __restrict__ emb,
                             unsigned short* __restrict__ xbf) {
  int idx = blockIdx.x * blockDim.x + threadIdx.x;
  if (idx < Bb * Tt * Ee) {
    int e  = idx & (Ee - 1);
    int bt = idx >> 7;                       // Ee == 128
    int tok = x[bt];
    xbf[idx] = f32_to_bf16(emb[(size_t)tok * Ee + e]);
  }
}

// ---------------------------------------------------------------------------
// Pack per-gate fp32 W(K,H) into combined bf16 W^T[4H, K] (+ combined bias)
// gate order i,f,o,g along N blocks of 256
// ---------------------------------------------------------------------------
__global__ void prep_weights_kernel(const float* __restrict__ Wi, const float* __restrict__ Wf,
                                    const float* __restrict__ Wo, const float* __restrict__ Wg,
                                    const float* __restrict__ bi, const float* __restrict__ bf_,
                                    const float* __restrict__ bo, const float* __restrict__ bg,
                                    unsigned short* __restrict__ WzT, float* __restrict__ biasc,
                                    int Ktot) {
  int idx = blockIdx.x * blockDim.x + threadIdx.x;
  int total = 4 * Hh * Ktot;
  if (idx < total) {
    int k   = idx % Ktot;
    int rem = idx / Ktot;
    int n   = rem & (Hh - 1);
    int g   = rem >> 8;                      // Hh == 256
    const float* W = (g == 0) ? Wi : (g == 1) ? Wf : (g == 2) ? Wo : Wg;
    WzT[idx] = f32_to_bf16(W[(size_t)k * Hh + n]);
  }
  if (idx < 4 * Hh) {
    int n = idx & (Hh - 1);
    int g = idx >> 8;
    const float* bv = (g == 0) ? bi : (g == 1) ? bf_ : (g == 2) ? bo : bg;
    biasc[idx] = bv[n];
  }
}

// ---------------------------------------------------------------------------
// Persistent BiLSTM layer: one 1024-thread block per direction (grid.x = 2).
// Each step: gates[32,1024] = [x_t | h] @ Wz  via v_wmma_f32_16x16x32_bf16.
// Wave w owns C tile (mtile = w&1, ncols nb..nb+15 of every gate block), so
// i/f/o/g for an element share lane+VGPR slot => cell update is register-only.
// h kept in LDS (bf16), c kept in VGPRs across all T steps.
// ---------------------------------------------------------------------------
__global__ __launch_bounds__(1024)
void lstm_layer_kernel(const unsigned short* __restrict__ Xbf, int in_dim,
                       const unsigned short* __restrict__ WzT_fw,
                       const unsigned short* __restrict__ WzT_bw,
                       const float* __restrict__ bias_fw,
                       const float* __restrict__ bias_bw,
                       float* __restrict__ Hout,                 // [B,T,2H] f32
                       unsigned short* __restrict__ Hout_bf,     // optional bf16 copy
                       const float* __restrict__ res)            // optional residual
{
  const int dir = blockIdx.x;
  const bool reverse = (dir == 1);
  const unsigned short* WzT  = reverse ? WzT_bw : WzT_fw;
  const float*          bsc  = reverse ? bias_bw : bias_fw;
  const int Ktot = in_dim + Hh;

  __shared__ __align__(16) unsigned short h_lds[Bb * Hh];
  for (int i = threadIdx.x; i < Bb * Hh; i += blockDim.x) h_lds[i] = 0;

  const int lane   = threadIdx.x & 31;
  const int w      = threadIdx.x >> 5;      // 32 waves
  const int mtile  = w & 1;                 // M tile (batch 0-15 / 16-31)
  const int nb     = (w >> 1) * 16;         // hidden-col base 0..240
  const int ln     = lane & 15;
  const int khalfA = (lane >> 4) * 8;       // A frag: lanes 0-15 K{0..7,16..23}, 16-31 K{8..15,24..31}
  const int koffB  = (lane >> 4) * 8;       // B frag: lanes 0-15 K{0..7,16..23}, 16-31 K{8..15,24..31}
  const int n_h    = nb + ln;

  const float bi_i = bsc[0 * Hh + n_h];
  const float bi_f = bsc[1 * Hh + n_h];
  const float bi_o = bsc[2 * Hh + n_h];
  const float bi_g = bsc[3 * Hh + n_h];

  const unsigned short* wrow0 = WzT + (size_t)(0 * Hh + n_h) * Ktot + koffB;
  const unsigned short* wrow1 = WzT + (size_t)(1 * Hh + n_h) * Ktot + koffB;
  const unsigned short* wrow2 = WzT + (size_t)(2 * Hh + n_h) * Ktot + koffB;
  const unsigned short* wrow3 = WzT + (size_t)(3 * Hh + n_h) * Ktot + koffB;

  const int arow = mtile * 16 + ln;         // batch row this lane supplies to A
  const unsigned short* xrow = Xbf + (size_t)arow * ((size_t)Tt * in_dim) + khalfA;
  const unsigned short* hrow = &h_lds[arow * Hh + khalfA];

  float creg[8];
  #pragma unroll
  for (int r = 0; r < 8; ++r) creg[r] = 0.f;

  const int Kx = in_dim >> 5;               // 32-wide K chunks from x (global)
  const int Kh = Hh >> 5;                   // 8 chunks from h (LDS)

  __syncthreads();

  for (int step = 0; step < Tt; ++step) {
    const int tt = reverse ? (Tt - 1 - step) : step;
    v8f a0 = {0,0,0,0,0,0,0,0};
    v8f a1 = a0, a2 = a0, a3 = a0;

    const unsigned short* xk = xrow + (size_t)tt * in_dim;
    for (int kc = 0; kc < Kx; ++kc) {
      const int kb = kc * 32;
      Frag16 af;
      af.q[0] = *(const u32x4*)(xk + kb);
      af.q[1] = *(const u32x4*)(xk + kb + 16);
      Frag16 b0v, b1v, b2v, b3v;
      b0v.q[0] = *(const u32x4*)(wrow0 + kb); b0v.q[1] = *(const u32x4*)(wrow0 + kb + 16);
      b1v.q[0] = *(const u32x4*)(wrow1 + kb); b1v.q[1] = *(const u32x4*)(wrow1 + kb + 16);
      b2v.q[0] = *(const u32x4*)(wrow2 + kb); b2v.q[1] = *(const u32x4*)(wrow2 + kb + 16);
      b3v.q[0] = *(const u32x4*)(wrow3 + kb); b3v.q[1] = *(const u32x4*)(wrow3 + kb + 16);
      a0 = __builtin_amdgcn_wmma_f32_16x16x32_bf16(false, af.v, false, b0v.v, (short)0, a0, false, false);
      a1 = __builtin_amdgcn_wmma_f32_16x16x32_bf16(false, af.v, false, b1v.v, (short)0, a1, false, false);
      a2 = __builtin_amdgcn_wmma_f32_16x16x32_bf16(false, af.v, false, b2v.v, (short)0, a2, false, false);
      a3 = __builtin_amdgcn_wmma_f32_16x16x32_bf16(false, af.v, false, b3v.v, (short)0, a3, false, false);
    }
    #pragma unroll
    for (int kc = 0; kc < Kh; ++kc) {
      const int kb = kc * 32;
      const int kw = in_dim + kb;
      Frag16 af;
      af.q[0] = *(const u32x4*)(hrow + kb);
      af.q[1] = *(const u32x4*)(hrow + kb + 16);
      Frag16 b0v, b1v, b2v, b3v;
      b0v.q[0] = *(const u32x4*)(wrow0 + kw); b0v.q[1] = *(const u32x4*)(wrow0 + kw + 16);
      b1v.q[0] = *(const u32x4*)(wrow1 + kw); b1v.q[1] = *(const u32x4*)(wrow1 + kw + 16);
      b2v.q[0] = *(const u32x4*)(wrow2 + kw); b2v.q[1] = *(const u32x4*)(wrow2 + kw + 16);
      b3v.q[0] = *(const u32x4*)(wrow3 + kw); b3v.q[1] = *(const u32x4*)(wrow3 + kw + 16);
      a0 = __builtin_amdgcn_wmma_f32_16x16x32_bf16(false, af.v, false, b0v.v, (short)0, a0, false, false);
      a1 = __builtin_amdgcn_wmma_f32_16x16x32_bf16(false, af.v, false, b1v.v, (short)0, a1, false, false);
      a2 = __builtin_amdgcn_wmma_f32_16x16x32_bf16(false, af.v, false, b2v.v, (short)0, a2, false, false);
      a3 = __builtin_amdgcn_wmma_f32_16x16x32_bf16(false, af.v, false, b3v.v, (short)0, a3, false, false);
    }
    __syncthreads();   // all reads of h_lds for this step complete

    #pragma unroll
    for (int r = 0; r < 8; ++r) {
      float gi = 1.f / (1.f + __expf(-(a0[r] + bi_i)));
      float gf = 1.f / (1.f + __expf(-(a1[r] + bi_f)));
      float go = 1.f / (1.f + __expf(-(a2[r] + bi_o)));
      float gg = tanhf(a3[r] + bi_g);
      float c  = gf * creg[r] + gi * gg;
      creg[r]  = c;
      float hv = go * tanhf(c);
      int m = mtile * 16 + (lane >> 4) * 8 + r;     // C/D layout: VGPR r, lane halves
      h_lds[m * Hh + n_h] = f32_to_bf16(hv);
      size_t oidx = ((size_t)m * Tt + tt) * (size_t)(2 * Hh) + (size_t)dir * Hh + n_h;
      float ov = res ? (hv + res[oidx]) : hv;
      Hout[oidx] = ov;
      if (Hout_bf) Hout_bf[oidx] = f32_to_bf16(ov);
    }
    __syncthreads();   // h writes visible before next step's reads
  }
}

// ---------------------------------------------------------------------------
// FC: emissions[m, 0..1] = out1[m, :512] @ fc_w + fc_b   (N=2, scalar VALU)
// ---------------------------------------------------------------------------
__global__ void fc_kernel(const float* __restrict__ X, const float* __restrict__ W,
                          const float* __restrict__ bv, float* __restrict__ em) {
  int m = blockIdx.x * blockDim.x + threadIdx.x;
  if (m < Bb * Tt) {
    const float* xr = X + (size_t)m * (2 * Hh);
    float s0 = bv[0], s1 = bv[1];
    for (int k = 0; k < 2 * Hh; ++k) {
      float xv = xr[k];
      s0 += xv * W[k * 2 + 0];
      s1 += xv * W[k * 2 + 1];
    }
    em[m * 2 + 0] = s0;
    em[m * 2 + 1] = s1;
  }
}

// ---------------------------------------------------------------------------
// CRF: gold score + forward algorithm (NT=2), one thread per batch, mean.
// ---------------------------------------------------------------------------
__device__ __forceinline__ float lse2(float a, float b) {
  float m = fmaxf(a, b);
  return m + __logf(__expf(a - m) + __expf(b - m));
}

__global__ void crf_kernel(const float* __restrict__ em, const int* __restrict__ tags,
                           const float* __restrict__ mask, const float* __restrict__ trans,
                           const float* __restrict__ start, const float* __restrict__ endv,
                           float* __restrict__ out) {
  __shared__ float red[Bb];
  int b = threadIdx.x;
  if (b < Bb) {
    const float* e  = em   + (size_t)b * Tt * NTAG;
    const int*   tg = tags + (size_t)b * Tt;
    const float* mk = mask + (size_t)b * Tt;
    float score = start[tg[0]];
    for (int t = 0; t < Tt - 1; ++t) {
      score += e[t * NTAG + tg[t]];
      score += trans[tg[t] * NTAG + tg[t + 1]] * mk[t + 1];
    }
    float msum = 0.f;
    for (int t = 0; t < Tt; ++t) msum += mk[t];
    int last = (int)msum - 1;
    score += endv[tg[last]];
    float a0 = start[0] + e[0], a1 = start[1] + e[1];
    for (int t = 1; t < Tt; ++t) {
      float e0 = e[t * NTAG + 0], e1 = e[t * NTAG + 1];
      float mt = mk[t];
      float n0 = lse2(a0 + e0 + trans[0], a1 + e1 + trans[2]);
      float n1 = lse2(a0 + e0 + trans[1], a1 + e1 + trans[3]);
      a0 = n0 * mt; a1 = n1 * mt;
    }
    float logz = lse2(a0 + endv[0], a1 + endv[1]);
    red[b] = logz - score;
  }
  __syncthreads();
  if (b == 0) {
    float s = 0.f;
    for (int i = 0; i < Bb; ++i) s += red[i];
    out[0] = s / (float)Bb;
  }
}

// ---------------------------------------------------------------------------
// Host launcher
// in order: x, tags, mask, embedding,
//           L0.fw{Wi,bi,Wf,bf,Wo,bo,Wg,bg}=4..11, L0.bw=12..19,
//           L1.fw=20..27, L1.bw=28..35, fc_w=36, fc_b=37,
//           crf_trans=38, crf_start=39, crf_end=40
// ---------------------------------------------------------------------------
extern "C" void kernel_launch(void* const* d_in, const int* in_sizes, int n_in,
                              void* d_out, int out_size, void* d_ws, size_t ws_size,
                              hipStream_t stream) {
  (void)in_sizes; (void)n_in; (void)out_size; (void)ws_size;
  const int*   x    = (const int*)d_in[0];
  const int*   tags = (const int*)d_in[1];
  const float* mask = (const float*)d_in[2];
  const float* emb  = (const float*)d_in[3];
  const float* fc_w = (const float*)d_in[36];
  const float* fc_b = (const float*)d_in[37];
  const float* ctr  = (const float*)d_in[38];
  const float* cst  = (const float*)d_in[39];
  const float* cen  = (const float*)d_in[40];

  char* ws = (char*)d_ws;
  size_t off = 0;
  auto alloc = [&](size_t bytes) -> char* {
    char* p = ws + off;
    off += (bytes + 255) & ~(size_t)255;
    return p;
  };

  const int K0 = Ee + Hh;        // 384
  const int K1 = 2 * Hh + Hh;    // 768

  unsigned short* xbf  = (unsigned short*)alloc((size_t)Bb * Tt * Ee * 2);
  float*          h0f  = (float*)         alloc((size_t)Bb * Tt * 2 * Hh * 4);
  unsigned short* h0b  = (unsigned short*)alloc((size_t)Bb * Tt * 2 * Hh * 2);
  float*          o1f  = (float*)         alloc((size_t)Bb * Tt * 2 * Hh * 4);
  float*          em   = (float*)         alloc((size_t)Bb * Tt * NTAG * 4);
  unsigned short* w0fw = (unsigned short*)alloc((size_t)4 * Hh * K0 * 2);
  unsigned short* w0bw = (unsigned short*)alloc((size_t)4 * Hh * K0 * 2);
  unsigned short* w1fw = (unsigned short*)alloc((size_t)4 * Hh * K1 * 2);
  unsigned short* w1bw = (unsigned short*)alloc((size_t)4 * Hh * K1 * 2);
  float*          b0fw = (float*)alloc((size_t)4 * Hh * 4);
  float*          b0bw = (float*)alloc((size_t)4 * Hh * 4);
  float*          b1fw = (float*)alloc((size_t)4 * Hh * 4);
  float*          b1bw = (float*)alloc((size_t)4 * Hh * 4);

  {
    int n = Bb * Tt * Ee;
    embed_kernel<<<(n + 255) / 256, 256, 0, stream>>>(x, emb, xbf);
  }

  auto P = [&](int base) { return base; };
  auto F = [&](int i) { return (const float*)d_in[i]; };
  // layer0 fw/bw (Ktot=384), layer1 fw/bw (Ktot=768)
  prep_weights_kernel<<<(4 * Hh * K0 + 255) / 256, 256, 0, stream>>>(
      F(P(4) + 0), F(P(4) + 2), F(P(4) + 4), F(P(4) + 6),
      F(P(4) + 1), F(P(4) + 3), F(P(4) + 5), F(P(4) + 7), w0fw, b0fw, K0);
  prep_weights_kernel<<<(4 * Hh * K0 + 255) / 256, 256, 0, stream>>>(
      F(P(12) + 0), F(P(12) + 2), F(P(12) + 4), F(P(12) + 6),
      F(P(12) + 1), F(P(12) + 3), F(P(12) + 5), F(P(12) + 7), w0bw, b0bw, K0);
  prep_weights_kernel<<<(4 * Hh * K1 + 255) / 256, 256, 0, stream>>>(
      F(P(20) + 0), F(P(20) + 2), F(P(20) + 4), F(P(20) + 6),
      F(P(20) + 1), F(P(20) + 3), F(P(20) + 5), F(P(20) + 7), w1fw, b1fw, K1);
  prep_weights_kernel<<<(4 * Hh * K1 + 255) / 256, 256, 0, stream>>>(
      F(P(28) + 0), F(P(28) + 2), F(P(28) + 4), F(P(28) + 6),
      F(P(28) + 1), F(P(28) + 3), F(P(28) + 5), F(P(28) + 7), w1bw, b1bw, K1);

  // Layer 0: in = embedding bf16, out = concat (f32 + bf16), no residual
  lstm_layer_kernel<<<2, 1024, 0, stream>>>(xbf, Ee, w0fw, w0bw, b0fw, b0bw,
                                            h0f, h0b, nullptr);
  // Layer 1: in = h0 bf16, out = concat + residual(h0 f32)
  lstm_layer_kernel<<<2, 1024, 0, stream>>>(h0b, 2 * Hh, w1fw, w1bw, b1fw, b1bw,
                                            o1f, nullptr, h0f);

  fc_kernel<<<(Bb * Tt + 255) / 256, 256, 0, stream>>>(o1f, fc_w, fc_b, em);
  crf_kernel<<<1, Bb, 0, stream>>>(em, tags, mask, ctr, cst, cen, (float*)d_out);
}
